// AttentionDotProduct_28372553957389
// MI455X (gfx1250) — matrix operand order repaired and checked by reading, not compiled, not run on hardware
//
#include <hip/hip_runtime.h>
#include <hip/hip_bf16.h>

// CDNA5 / gfx1250 attention forward: out = softmax(mask(Q K^T))·V, plus full weights.
// - bf16 WMMA (v_wmma_f32_16x16x32_bf16) for both GEMMs, f32 accumulate.
// - K tiles streamed global->LDS with GLOBAL_LOAD_ASYNC_TO_LDS_B128 (ASYNCcnt),
//   double-buffered per wave so DMA overlaps the WMMAs.
// - 16-query tile per workgroup; full 16x2048 score row lives in LDS (128KB),
//   total ~260KB dynamic LDS -- legal only on CDNA5's 320KB/WGP.

typedef __attribute__((ext_vector_type(16))) __bf16 v16bf;
typedef __attribute__((ext_vector_type(8)))  float  v8f;

#define B_   8
#define S_   2048
#define D_   128
#define QT   16          // query rows per workgroup (WMMA M)
#define NW   8           // waves per workgroup (wave32 -> 256 threads)
#define SCALE 0.08838834764831845f   // 1/sqrt(128)
#define NEGH  (-1.0e9f)

// dynamic LDS layout (bytes):
//   [0, 131072)            float  scoreS[QT][S_]
//   [131072, 135168)       __bf16 Qs[QT][D_]
//   [135168, 266240)       float  Kf[NW][2][16][D_]   (double-buffered async staging)
#define LDS_BYTES (QT*S_*4 + QT*D_*2 + NW*2*16*D_*4)

// Issue one 16x128-f32 tile (8KB) as 16 async B128 DMA ops (32 lanes x 16B each).
// Tracked by ASYNCcnt; caller waits with s_wait_asynccnt.
__device__ __forceinline__ void async_stage_tile(const float* __restrict__ gsrc,
                                                 float* ldst, int lane)
{
    #pragma unroll
    for (int i = 0; i < 16; ++i) {
        const int idx = i*32 + lane;                      // float4 index
        const unsigned lds = (unsigned)(size_t)(ldst + 4*idx);  // LDS byte address (addr[31:0])
        const float*  g   = gsrc + 4*idx;
        asm volatile("global_load_async_to_lds_b128 %0, %1, off"
                     :: "v"(lds), "v"(g) : "memory");
    }
}

__global__ __launch_bounds__(256)
void attn_fwd_wmma(const float* __restrict__ Q, const float* __restrict__ K,
                   const float* __restrict__ V, const int* __restrict__ Mpad,
                   float* __restrict__ outV, float* __restrict__ outW)
{
    extern __shared__ char smem[];
    float*  scoreS = (float*)smem;                            // [QT][S_]
    __bf16* Qs     = (__bf16*)(smem + QT*S_*4);               // [QT][D_]
    float*  KfAll  = (float*)(smem + QT*S_*4 + QT*D_*2);      // [NW][2][16][D_]

    const int b    = blockIdx.x / (S_/QT);
    const int qt   = blockIdx.x % (S_/QT);
    const int q0   = qt * QT;
    const int tid  = threadIdx.x;
    const int w    = tid >> 5;        // wave id (wave32)
    const int lane = tid & 31;
    const int hi   = lane >> 4;       // lane half (ISA fragment layouts)
    const int l16  = lane & 15;

    const int numKT   = qt + 1;       // causal: key tiles 0..qt
    const int numCols = numKT * QT;

    // ---------------- stage Q tile -> LDS bf16 ----------------
    const float* Qg = Q + ((size_t)b*S_ + q0) * D_;
    for (int i = tid; i < QT*D_; i += 256)
        Qs[i] = (__bf16)Qg[i];
    __syncthreads();

    // A fragments for Q (16x32 bf16 layout, 4 chunks across D=128)
    v16bf afrag[4];
    #pragma unroll
    for (int c = 0; c < 4; ++c) {
        #pragma unroll
        for (int vv = 0; vv < 8; ++vv) {
            const int kl = (vv & 3)*2 + hi*8 + (vv >> 2)*16;   // K index in chunk
            afrag[c][2*vv+0] = Qs[l16*D_ + 32*c + kl + 0];
            afrag[c][2*vv+1] = Qs[l16*D_ + 32*c + kl + 1];
        }
    }

    // ---------------- scores = mask(Q K^T) * scale -> LDS ----------------
    // Key tiles round-robin over waves; per-wave double-buffered async staging.
    float* Kw = KfAll + w*(2*16*D_);           // this wave's two buffers
    const float* Kb = K + (size_t)b*S_*D_;
    if (w < numKT)
        async_stage_tile(Kb + (size_t)(w*QT)*D_, Kw, lane);   // prologue: buffer 0
    int bufIdx = 0;
    for (int kt = w; kt < numKT; kt += NW) {
        const int ktNext = kt + NW;
        if (ktNext < numKT)                     // stream next tile into other buffer
            async_stage_tile(Kb + (size_t)(ktNext*QT)*D_, Kw + (1-bufIdx)*(16*D_), lane);
        // wait for *current* tile (leave the 16 next-tile DMAs in flight)
        if (ktNext < numKT)
            asm volatile("s_wait_asynccnt 0x10" ::: "memory");
        else
            asm volatile("s_wait_asynccnt 0x0"  ::: "memory");

        const float* Kf = Kw + bufIdx*(16*D_);  // staged f32 K tile [16][D_]
        v8f acc = {};
        #pragma unroll
        for (int c = 0; c < 4; ++c) {
            v16bf bfrag;                        // B = K^T tile (32x16 layout)
            #pragma unroll
            for (int vv = 0; vv < 8; ++vv) {
                const int kd = (vv & 3)*2 + hi*8 + (vv >> 2)*16;    // K index in chunk
                const float2 f2 = *(const float2*)(Kf + l16*D_ + 32*c + kd);
                bfrag[2*vv+0] = (__bf16)f2.x;
                bfrag[2*vv+1] = (__bf16)f2.y;
            }
            acc = __builtin_amdgcn_wmma_f32_16x16x32_bf16(
                      false, afrag[c], false, bfrag, (short)0, acc, false, false);
        }
        // mask (causal + padding, pre-scale NEG_HUGE like reference), store
        const int k0  = kt * QT;
        const int key = k0 + l16;
        const int pad = Mpad[b*S_ + key];
        #pragma unroll
        for (int vv = 0; vv < 8; ++vv) {
            const int m    = vv + hi*8;                 // C-frag row
            const int qrow = q0 + m;
            const float s  = ((key <= qrow) && (pad == 0)) ? acc[vv] : NEGH;
            scoreS[m*S_ + key] = s * SCALE;
        }
        bufIdx ^= 1;
    }
    __syncthreads();

    // ---------------- softmax (2 rows per wave) + write weights ----------------
    float* Wg = outW + ((size_t)b*S_ + q0) * S_;
    for (int r = 2*w; r < 2*w + 2; ++r) {
        float* row = scoreS + r*S_;
        float mx = -3.4e38f;
        for (int c = lane; c < numCols; c += 32)
            mx = fmaxf(mx, row[c]);
        #pragma unroll
        for (int off = 16; off; off >>= 1)
            mx = fmaxf(mx, __shfl_xor(mx, off, 32));
        float sum = 0.f;
        for (int c = lane; c < numCols; c += 32)
            sum += __expf(row[c] - mx);
        #pragma unroll
        for (int off = 16; off; off >>= 1)
            sum += __shfl_xor(sum, off, 32);
        const float inv = 1.0f / sum;
        for (int c = lane; c < numCols; c += 32) {
            const float p = __expf(row[c] - mx) * inv;
            row[c] = p;                        // keep for PV GEMM
            Wg[(size_t)r*S_ + c] = p;          // emit weight
        }
        for (int c = numCols + lane; c < S_; c += 32)
            Wg[(size_t)r*S_ + c] = 0.0f;       // causal tail: exactly 0
    }
    __syncthreads();

    // ---------------- out = P · V  (wave w owns d columns [16w,16w+16)) ----------------
    const int d0  = 16 * w;
    const int nCh = (numCols + 31) >> 5;       // 32-key chunks
    const float* Vg = V + ((size_t)b*S_) * D_;
    v8f oacc = {};
    for (int ch = 0; ch < nCh; ++ch) {
        const int kbase = ch * 32;
        if (kbase + 32 < S_)                   // prefetch next V chunk
            __builtin_prefetch(Vg + (size_t)(kbase + 32)*D_ + d0, 0, 1);
        v16bf pfrag, vfrag;
        #pragma unroll
        for (int vv = 0; vv < 8; ++vv) {       // A = P tile (16x32)
            const int kl   = (vv & 3)*2 + hi*8 + (vv >> 2)*16;
            const int key0 = kbase + kl;
            pfrag[2*vv+0] = (key0 + 0 < numCols) ? (__bf16)scoreS[l16*S_ + key0 + 0] : (__bf16)0.f;
            pfrag[2*vv+1] = (key0 + 1 < numCols) ? (__bf16)scoreS[l16*S_ + key0 + 1] : (__bf16)0.f;
        }
        #pragma unroll
        for (int vv = 0; vv < 8; ++vv) {       // B = V tile (32x16)
            const int kd = kbase + (vv & 3)*2 + hi*8 + (vv >> 2)*16;
            vfrag[2*vv+0] = (__bf16)Vg[(size_t)(kd + 0)*D_ + d0 + l16];
            vfrag[2*vv+1] = (__bf16)Vg[(size_t)(kd + 1)*D_ + d0 + l16];
        }
        oacc = __builtin_amdgcn_wmma_f32_16x16x32_bf16(
                   false, pfrag, false, vfrag, (short)0, oacc, false, false);
    }
    float* Og = outV + ((size_t)b*S_ + q0) * D_;
    #pragma unroll
    for (int vv = 0; vv < 8; ++vv) {
        const int m = vv + hi*8;
        Og[(size_t)m*D_ + d0 + l16] = oacc[vv];
    }
}

extern "C" void kernel_launch(void* const* d_in, const int* in_sizes, int n_in,
                              void* d_out, int out_size, void* d_ws, size_t ws_size,
                              hipStream_t stream) {
    (void)in_sizes; (void)n_in; (void)out_size; (void)d_ws; (void)ws_size;
    const float* Q    = (const float*)d_in[0];
    const float* K    = (const float*)d_in[1];
    const float* V    = (const float*)d_in[2];
    const int*   Mpad = (const int*)d_in[3];
    float* outV = (float*)d_out;                          // [B,S,D]
    float* outW = (float*)d_out + (size_t)B_*S_*D_;       // [B,S,S]

    // ~260KB dynamic LDS: legal on CDNA5 (320KB/WGP); raise the runtime cap.
    (void)hipFuncSetAttribute((const void*)attn_fwd_wmma,
                              hipFuncAttributeMaxDynamicSharedMemorySize, LDS_BYTES);

    dim3 grid(B_ * (S_/QT));   // 1024 workgroups
    dim3 block(256);           // 8 waves (wave32)
    attn_fwd_wmma<<<grid, block, LDS_BYTES, stream>>>(Q, K, V, Mpad, outV, outW);
}
